// Model_36962488550132
// MI455X (gfx1250) — matrix operand order repaired and checked by reading, not compiled, not run on hardware
//
#include <hip/hip_runtime.h>
#include <stdint.h>
#include <math.h>

// ---------------------------------------------------------------------------
// Strategy: the reference only consumes 690 rfft bins per row. Computing those
// bins directly is a GEMM  X(8192x8192) @ B(8192x1380)  -> WMMA bf16 on CDNA5.
// Precision: bf16x3 split (xh*Bh + xl*Bh + xh*Bl), fp32 accumulate -> ~1e-5.
// Data movement: triple-buffered GLOBAL_LOAD_ASYNC_TO_LDS_B128 (ASYNCcnt),
// N-fast grid so each 6.3MB A stripe is reused by all 11 column blocks in L2.
// Tile: block 128x128, 8 waves of 32x64 -> 8 wmma per 12 ds_load_b128.
// ---------------------------------------------------------------------------

typedef __bf16 bf16;
typedef __bf16 v16bf __attribute__((ext_vector_type(16)));
typedef float  v8f   __attribute__((ext_vector_type(8)));

#define K_TOT 8192
#define NCOL  1408        // 1380 real cos/sin columns, padded to 11*128
#define BM    128
#define BN    128
#define BK    64
#define NBUF  3
#define PITCH 72          // LDS row pitch in bf16 elems: 144B, 16B-aligned, bank-spread

__device__ __forceinline__ uint32_t lds_off(const void* p) {
  // generic LDS pointer: low 32 bits are the LDS byte offset
  return (uint32_t)(uintptr_t)p;
}

__device__ __forceinline__ void async_b128(uint32_t lds, const void* gbase, uint32_t voff) {
  // GLOBAL_LOAD_ASYNC_TO_LDS_B128, GVS mode: mem = SADDR + VADDR(32b offset)
  asm volatile("global_load_async_to_lds_b128 %0, %1, %2"
               :: "v"(lds), "v"(voff), "s"(gbase) : "memory");
}

__device__ __forceinline__ float gelu_exact(float x) {
  return 0.5f * x * (1.0f + erff(x * 0.70710678118654752f));
}

union FragBF { struct { uint4 lo, hi; } u; v16bf v; };

// ---------------------------------------------------------------------------
// Kernel 1: split fp32 x into bf16 hi + bf16 lo residual
// ---------------------------------------------------------------------------
__global__ __launch_bounds__(256)
void split_bf16_kernel(const float* __restrict__ x, bf16* __restrict__ xh,
                       bf16* __restrict__ xl) {
  size_t i = ((size_t)blockIdx.x * 256 + threadIdx.x) * 4;
  float4 v = *(const float4*)(x + i);
  bf16 h[4], l[4];
  float f[4] = {v.x, v.y, v.z, v.w};
#pragma unroll
  for (int j = 0; j < 4; ++j) {
    h[j] = (bf16)f[j];
    l[j] = (bf16)(f[j] - (float)h[j]);
  }
  *(uint2*)(xh + i) = *(uint2*)h;
  *(uint2*)(xl + i) = *(uint2*)l;
}

// ---------------------------------------------------------------------------
// Kernel 2: basis B^T[j][k], j = 2*bin (cos) / 2*bin+1 (sin), bf16 hi/lo.
// Integer (f*k) mod 8192 keeps the cos/sin argument small and exact.
// ---------------------------------------------------------------------------
__global__ __launch_bounds__(256)
void build_basis_kernel(bf16* __restrict__ BhT, bf16* __restrict__ BlT) {
  int j = blockIdx.x;             // 0..1407
  int b = j >> 1, is_sin = j & 1;
  bool valid = (j < 1380);
  int f;
  if      (b < 30)  f = 22 + b;        // branch1: bins 22..51
  else if (b < 90)  f = 13 + b;        // branch2: bins 43..102
  else if (b < 210) f = b - 4;         // branch3: bins 86..205
  else if (b < 450) f = b - 38;        // branch4: bins 172..411
  else              f = 2 * b - 556;   // branch5: bins 344,346,..,822
  const float w = 7.66990393942820605e-4f; // 2*pi/8192
  for (int k = threadIdx.x; k < K_TOT; k += 256) {
    float val = 0.0f;
    if (valid) {
      int m = (f * k) & 8191;
      float ang = (float)m * w;
      val = is_sin ? sinf(ang) : cosf(ang);
    }
    bf16 h = (bf16)val;
    bf16 l = (bf16)(val - (float)h);
    BhT[(size_t)j * K_TOT + k] = h;
    BlT[(size_t)j * K_TOT + k] = l;
  }
}

// ---------------------------------------------------------------------------
// Kernel 3: WMMA GEMM over K' = 3*8192 (xh*Bh, xl*Bh, xh*Bl) with fused
// |re,im| -> XA_T[bin][row].  Block tile 128x128, 8 waves of 32x64,
// triple-buffered async-to-LDS pipeline.
// ---------------------------------------------------------------------------
__global__ __launch_bounds__(256)
void dft_gemm_abs_kernel(const bf16* __restrict__ xh, const bf16* __restrict__ xl,
                         const bf16* __restrict__ BhT, const bf16* __restrict__ BlT,
                         float* __restrict__ xaT) {
  __shared__ bf16 As[NBUF][BM * PITCH];
  __shared__ bf16 Bs[NBUF][BN * PITCH];

  const int tid  = threadIdx.x;
  const int wave = tid >> 5, lane = tid & 31;
  const int ln = lane & 15, lh = lane >> 4;
  const int n0 = blockIdx.x * BN;     // column block (N fast -> L2 A-stripe reuse)
  const int m0 = blockIdx.y * BM;
  const int mw = (wave >> 1) * 32;    // 4 waves tile M
  const int nw = (wave & 1) * 64;     // 2 waves tile N

  v8f acc[2][4] = {};

  auto issue = [&](int stage, int buf) {
    const int kg  = stage * BK;
    const int seg = kg >> 13;               // 0: xh*Bh  1: xl*Bh  2: xh*Bl
    const uint32_t kin = (uint32_t)(kg & (K_TOT - 1));
    const bf16* Ag = (seg == 1) ? xl : xh;
    const bf16* Bg = (seg == 2) ? BlT : BhT;
#pragma unroll
    for (int i = 0; i < 4; ++i) {           // A: 128 rows x 8 chunks of 16B
      int c = tid + 256 * i;
      int m = c >> 3, ch = c & 7;
      uint32_t l = lds_off(&As[buf][m * PITCH + ch * 8]);
      uint32_t g = (((uint32_t)(m0 + m)) * (uint32_t)K_TOT + kin + (uint32_t)(ch * 8)) * 2u;
      async_b128(l, Ag, g);
    }
#pragma unroll
    for (int i = 0; i < 4; ++i) {           // B: 128 rows x 8 chunks of 16B
      int c = tid + 256 * i;
      int n = c >> 3, ch = c & 7;
      uint32_t l = lds_off(&Bs[buf][n * PITCH + ch * 8]);
      uint32_t g = (((uint32_t)(n0 + n)) * (uint32_t)K_TOT + kin + (uint32_t)(ch * 8)) * 2u;
      async_b128(l, Bg, g);
    }
  };

  const int S = (3 * K_TOT) / BK;           // 384 stages
  issue(0, 0);
  issue(1, 1);
  for (int s = 0; s < S; ++s) {
    int buf = s % NBUF;
    if (s + 2 < S) {
      issue(s + 2, (s + 2) % NBUF);
      asm volatile("s_wait_asynccnt 16" ::: "memory");  // 2 stages (8 instr each) in flight
    } else if (s + 1 < S) {
      asm volatile("s_wait_asynccnt 8" ::: "memory");   // 1 stage in flight
    } else {
      asm volatile("s_wait_asynccnt 0" ::: "memory");
    }
    __syncthreads();
#pragma unroll
    for (int ks = 0; ks < 2; ++ks) {
      FragBF a[2], b[4];
#pragma unroll
      for (int i = 0; i < 2; ++i) {         // A 16x32 frag: e0-7 = K 8h..8h+7, e8-15 = K 16+8h..
        int m = mw + i * 16 + ln;
        const bf16* p = &As[buf][m * PITCH + ks * 32 + 8 * lh];
        a[i].u.lo = *(const uint4*)p;
        a[i].u.hi = *(const uint4*)(p + 16);
      }
#pragma unroll
      for (int j = 0; j < 4; ++j) {         // B 32x16 frag: e0-7 = K 8h..8h+7, e8-15 = K 16+8h.. of col n
        int n = nw + j * 16 + ln;
        const bf16* p = &Bs[buf][n * PITCH + ks * 32 + 8 * lh];
        b[j].u.lo = *(const uint4*)p;
        b[j].u.hi = *(const uint4*)(p + 16);
      }
#pragma unroll
      for (int i = 0; i < 2; ++i)
#pragma unroll
        for (int j = 0; j < 4; ++j)
          acc[i][j] = __builtin_amdgcn_wmma_f32_16x16x32_bf16(
              false, a[i].v, false, b[j].v, (short)0, acc[i][j], false, false);
    }
    __syncthreads();
  }

  // Epilogue: pair cos(even n)/sin(odd n) via lane xor-1, write |.| bin-major.
#pragma unroll
  for (int i = 0; i < 2; ++i)
#pragma unroll
    for (int j = 0; j < 4; ++j) {
      int n_g = n0 + nw + j * 16 + ln;
#pragma unroll
      for (int v = 0; v < 8; ++v) {
        float re = acc[i][j][v];
        float im = __shfl_xor(re, 1, 32);
        if ((lane & 1) == 0) {
          int m_g = m0 + mw + i * 16 + v + 8 * lh;
          xaT[(size_t)(n_g >> 1) * K_TOT + m_g] = sqrtf(re * re + im * im);
        }
      }
    }
}

// ---------------------------------------------------------------------------
// Kernel 4: 5 conv branches (a:stride sa + GELU, b:k3 + GELU, c:k2 s2 + GELU),
// one thread per (row, branch), weights in LDS, writes concat transposed.
// ---------------------------------------------------------------------------
struct ConvParams {
  const float* wa[5]; const float* ba[5];
  const float* wb[5]; const float* bb[5];
  const float* wc[5]; const float* bc[5];
};

__global__ __launch_bounds__(256)
void conv_branches_kernel(const float* __restrict__ xaT, float* __restrict__ hT,
                          ConvParams P) {
  __shared__ float swa[4 * 34], sba[4], swb[192], sbb[16], swc[512], sbc[16];
  const int br  = blockIdx.y;
  const int KA[5]  = {5, 9, 17, 34, 34};
  const int SA[5]  = {1, 2, 4, 8, 8};
  const int OFF[5] = {0, 30, 90, 210, 450};
  const int ka = KA[br], sa = SA[br], off = OFF[br];
  const int tid = threadIdx.x;
  for (int i = tid; i < 4 * ka; i += 256) swa[i] = P.wa[br][i];
  for (int i = tid; i < 4;      i += 256) sba[i] = P.ba[br][i];
  for (int i = tid; i < 192;    i += 256) swb[i] = P.wb[br][i];
  for (int i = tid; i < 16;     i += 256) sbb[i] = P.bb[br][i];
  for (int i = tid; i < 512;    i += 256) swc[i] = P.wc[br][i];
  for (int i = tid; i < 16;     i += 256) sbc[i] = P.bc[br][i];
  __syncthreads();

  const int row = blockIdx.x * 256 + tid;
  const float* xcol = xaT + (size_t)off * K_TOT + row;  // bin-major -> coalesced

  for (int t = 0; t < 12; ++t) {
    float av[4][4];
#pragma unroll
    for (int q = 0; q < 4; ++q) {           // conv-a positions 2t..2t+3
      int p = 2 * t + q;
      float s0 = sba[0], s1 = sba[1], s2 = sba[2], s3 = sba[3];
      int base = p * sa;
      for (int u = 0; u < ka; ++u) {
        float xv = xcol[(size_t)(base + u) * K_TOT];
        s0 += swa[0 * ka + u] * xv;
        s1 += swa[1 * ka + u] * xv;
        s2 += swa[2 * ka + u] * xv;
        s3 += swa[3 * ka + u] * xv;
      }
      av[q][0] = gelu_exact(s0); av[q][1] = gelu_exact(s1);
      av[q][2] = gelu_exact(s2); av[q][3] = gelu_exact(s3);
    }
    float bv[2][16];
#pragma unroll
    for (int pp = 0; pp < 2; ++pp)
      for (int co = 0; co < 16; ++co) {
        float s = sbb[co];
#pragma unroll
        for (int ci = 0; ci < 4; ++ci)
#pragma unroll
          for (int u = 0; u < 3; ++u)
            s += swb[(co * 4 + ci) * 3 + u] * av[pp + u][ci];
        bv[pp][co] = gelu_exact(s);
      }
    for (int co = 0; co < 16; ++co) {
      float s = sbc[co];
#pragma unroll
      for (int ci = 0; ci < 16; ++ci)
        s += swc[(co * 16 + ci) * 2 + 0] * bv[0][ci]
           + swc[(co * 16 + ci) * 2 + 1] * bv[1][ci];
      hT[(size_t)(co * 60 + br * 12 + t) * K_TOT + row] = gelu_exact(s);
    }
  }
}

// ---------------------------------------------------------------------------
// Kernel 5: out = sigmoid(h @ lw.T + lb)
// ---------------------------------------------------------------------------
__global__ __launch_bounds__(256)
void linear_sigmoid_kernel(const float* __restrict__ hT, const float* __restrict__ lw,
                           const float* __restrict__ lb, float* __restrict__ out) {
  __shared__ float sw[12 * 960];
  __shared__ float sb[12];
  const int tid = threadIdx.x;
  for (int i = tid; i < 12 * 960; i += 256) sw[i] = lw[i];
  if (tid < 12) sb[tid] = lb[tid];
  __syncthreads();
  const int row = blockIdx.x * 256 + tid;
  float acc[12];
#pragma unroll
  for (int j = 0; j < 12; ++j) acc[j] = sb[j];
  for (int i = 0; i < 960; ++i) {
    float h = hT[(size_t)i * K_TOT + row];
#pragma unroll
    for (int j = 0; j < 12; ++j) acc[j] += sw[j * 960 + i] * h;
  }
#pragma unroll
  for (int j = 0; j < 12; ++j)
    out[(size_t)row * 12 + j] = 1.0f / (1.0f + expf(-acc[j]));
}

// ---------------------------------------------------------------------------
extern "C" void kernel_launch(void* const* d_in, const int* in_sizes, int n_in,
                              void* d_out, int out_size, void* d_ws, size_t ws_size,
                              hipStream_t stream) {
  (void)in_sizes; (void)n_in; (void)out_size; (void)ws_size;
  const float* x = (const float*)d_in[0];

  char* w = (char*)d_ws;                                   // ~352 MB total
  bf16* xh  = (bf16*)w;  w += (size_t)K_TOT * K_TOT * 2;   // 128 MB
  bf16* xl  = (bf16*)w;  w += (size_t)K_TOT * K_TOT * 2;   // 128 MB
  bf16* BhT = (bf16*)w;  w += (size_t)NCOL * K_TOT * 2;    // 22.6 MB
  bf16* BlT = (bf16*)w;  w += (size_t)NCOL * K_TOT * 2;    // 22.6 MB
  float* xaT = (float*)w; w += (size_t)(NCOL / 2) * K_TOT * 4; // 23 MB (704 bins)
  float* hT  = (float*)w; w += (size_t)960 * K_TOT * 4;    // 31.5 MB

  split_bf16_kernel<<<65536, 256, 0, stream>>>(x, xh, xl);
  build_basis_kernel<<<NCOL, 256, 0, stream>>>(BhT, BlT);
  dft_gemm_abs_kernel<<<dim3(NCOL / BN, K_TOT / BM), 256, 0, stream>>>(xh, xl, BhT, BlT, xaT);

  ConvParams P;
  for (int b = 0; b < 5; ++b) {
    P.wa[b] = (const float*)d_in[1 + b * 6 + 0];
    P.ba[b] = (const float*)d_in[1 + b * 6 + 1];
    P.wb[b] = (const float*)d_in[1 + b * 6 + 2];
    P.bb[b] = (const float*)d_in[1 + b * 6 + 3];
    P.wc[b] = (const float*)d_in[1 + b * 6 + 4];
    P.bc[b] = (const float*)d_in[1 + b * 6 + 5];
  }
  conv_branches_kernel<<<dim3(32, 5), 256, 0, stream>>>(xaT, hT, P);
  linear_sigmoid_kernel<<<32, 256, 0, stream>>>(hT, (const float*)d_in[31],
                                                (const float*)d_in[32], (float*)d_out);
}
